// SelfAttention_37821482009201
// MI455X (gfx1250) — matrix operand order repaired and checked by reading, not compiled, not run on hardware
//
#include <hip/hip_runtime.h>

// ---------------------------------------------------------------------------
// Self-attention with per-head tanh-MLP QKV, flash-attention softmax, and
// LayerNorm, for MI455X (gfx1250). bf16 WMMA 16x16x32 with f32 accumulation.
// Software-pipelined attention: async global->LDS (ASYNCcnt) double-buffered
// K tiles overlap with WMMA compute; V tile loads hoisted into registers.
//   B=8, H=16, D=64, S=1024, E=1024
// ---------------------------------------------------------------------------

#define H_ 16
#define D_ 64
#define B_ 8
#define S_ 1024
#define E_ 1024

typedef unsigned short u16;
typedef unsigned int   u32;
typedef unsigned long long u64;

typedef __attribute__((ext_vector_type(8)))  __bf16 v8bf;
typedef __attribute__((ext_vector_type(16))) __bf16 v16bf;
typedef __attribute__((ext_vector_type(8)))  float  v8f;

union bfu { u16 u; __bf16 b; };
__device__ __forceinline__ __bf16 u2b(u16 u) { bfu t; t.u = u; return t.b; }
__device__ __forceinline__ u16 f2bu(float f) {
  u32 x = __builtin_bit_cast(u32, f);
  x += 0x7fffu + ((x >> 16) & 1u);          // round-to-nearest-even
  return (u16)(x >> 16);
}
__device__ __forceinline__ __bf16 f2b(float f) { return u2b(f2bu(f)); }

__device__ __forceinline__ v8f zero8() {
  v8f z = {0.f,0.f,0.f,0.f,0.f,0.f,0.f,0.f};
  return z;
}
__device__ __forceinline__ v16bf cat16(v8bf lo, v8bf hi) {
  return __builtin_shufflevector(lo, hi, 0,1,2,3,4,5,6,7,8,9,10,11,12,13,14,15);
}

// Branch-free tanh: tanh(x) = 1 - 2/(exp2(2*log2e*x)+1).
__device__ __forceinline__ float fast_tanh(float x) {
  float e = __builtin_amdgcn_exp2f(2.8853900818f * x);
  float r = __builtin_amdgcn_rcpf(e + 1.0f);
  return __builtin_fmaf(-2.0f, r, 1.0f);
}

// A fragment, 16-bit A 16x32 (ISA 7.12.2): lane m (both halves) holds row M=m;
// elems 0..7 -> K = half*8 + j, elems 8..15 -> K = 16 + half*8 + j.
__device__ __forceinline__ v16bf loadA(const __bf16* row, int kbase) {
  v8bf lo = *(const v8bf*)(row + kbase);
  v8bf hi = *(const v8bf*)(row + kbase + 16);
  return cat16(lo, hi);
}
// B fragment, 16-bit B 32x16, tile stored n-major (Bt[n][k]): lane n=lane&15,
// elems 0..7 -> K = half*8 + j, elems 8..15 -> K = 16 + half*8 + j.  kbase = kc*32 + half*8.
__device__ __forceinline__ v16bf loadB(const __bf16* nrow, int kbase) {
  v8bf lo = *(const v8bf*)(nrow + kbase);
  v8bf hi = *(const v8bf*)(nrow + kbase + 16);
  return cat16(lo, hi);
}

__device__ __forceinline__ v8f wmma_bf16(v16bf a, v16bf b, v8f c) {
  return __builtin_amdgcn_wmma_f32_16x16x32_bf16(false, a, false, b,
                                                 (short)0, c, false, false);
}

// --- gfx1250 async global->LDS copy (ASYNCcnt path) ---
__device__ __forceinline__ u32 lds_off(const void* p) {
  return (u32)(uintptr_t)p;       // flat LDS aperture: addr[31:0] = LDS offset
}
__device__ __forceinline__ void async_ld_b128(u32 dst_lds, const void* src) {
  asm volatile("global_load_async_to_lds_b128 %0, %1, off"
               :: "v"(dst_lds), "v"((u64)(uintptr_t)src)
               : "memory");
}
__device__ __forceinline__ void wait_async0() {
  asm volatile("s_wait_asynccnt 0" ::: "memory");
}

// One 64->64 dense layer with tanh epilogue for one wave's 16 rows.
template <bool TO_LDS>
__device__ __forceinline__ void gemm_layer(const __bf16* in_tile,
                                           const __bf16* wt,
                                           const float*  bias64,
                                           __bf16* out_lds,   // [128][72]
                                           u16*    out_glb,   // row stride 64
                                           int wv, int lane) {
  const int n16 = lane & 15, half = lane >> 4;
  const __bf16* arow = in_tile + (wv * 16 + n16) * 72;
  // A fragments invariant across the 4 output col-tiles: hoist into registers
  v16bf a0 = loadA(arow, half * 8);
  v16bf a1 = loadA(arow, 32 + half * 8);
#pragma unroll
  for (int nd = 0; nd < 4; ++nd) {
    const __bf16* wrow = wt + (nd * 16 + n16) * 72;
    v8f acc = zero8();
    acc = wmma_bf16(a0, loadB(wrow, half * 8), acc);
    acc = wmma_bf16(a1, loadB(wrow, 32 + half * 8), acc);
    float bv = bias64[nd * 16 + n16];
#pragma unroll
    for (int r = 0; r < 8; ++r) {
      float t = fast_tanh(acc[r] + bv);
      int row = wv * 16 + r + half * 8;
      int col = nd * 16 + n16;
      if constexpr (TO_LDS) out_lds[row * 72 + col] = f2b(t);
      else                  out_glb[(size_t)row * D_ + col] = f2bu(t);
    }
  }
}

// ---------------------------------------------------------------------------
// Kernel 1: xh = x + posenc;  Q,K,V = tanh(tanh(xh W1 + b1) W2 + b2)
// ---------------------------------------------------------------------------
__global__ __launch_bounds__(256) void qkv_posenc_kernel(
    const float* __restrict__ x,
    const float* __restrict__ Wq1, const float* __restrict__ bq1,
    const float* __restrict__ Wq2, const float* __restrict__ bq2,
    const float* __restrict__ Wk1, const float* __restrict__ bk1,
    const float* __restrict__ Wk2, const float* __restrict__ bk2,
    const float* __restrict__ Wv1, const float* __restrict__ bv1,
    const float* __restrict__ Wv2, const float* __restrict__ bv2,
    float* __restrict__ xh_ws, u16* __restrict__ Qg,
    u16* __restrict__ Kg, u16* __restrict__ Vg) {
  __shared__ alignas(16) __bf16 Wt[6][64][72];   // transposed weights (n-major)
  __shared__ float bias6[6][64];
  __shared__ float divt[512];                    // posenc divisor table
  __shared__ alignas(16) __bf16 xt[128][72];     // xh tile, bf16
  __shared__ alignas(16) __bf16 zt[128][72];     // hidden tile

  const int tid = threadIdx.x;
  const int bid = blockIdx.x;
  const int h  = bid >> 6;
  const int b  = (bid >> 3) & 7;
  const int s0 = (bid & 7) << 7;

  const float* Wsrc[6] = {Wq1 + h * 4096, Wq2 + h * 4096, Wk1 + h * 4096,
                          Wk2 + h * 4096, Wv1 + h * 4096, Wv2 + h * 4096};
  const float* bsrc[6] = {bq1 + h * 64, bq2 + h * 64, bk1 + h * 64,
                          bk2 + h * 64, bv1 + h * 64, bv2 + h * 64};

  for (int idx = tid; idx < 6 * 4096; idx += 256) {
    int w6 = idx >> 12, rem = idx & 4095;
    int dd = rem >> 6, e = rem & 63;
    Wt[w6][e][dd] = f2b(Wsrc[w6][rem]);          // store transposed
  }
  for (int idx = tid; idx < 6 * 64; idx += 256)
    bias6[idx >> 6][idx & 63] = bsrc[idx >> 6][idx & 63];
  for (int idx = tid; idx < 512; idx += 256)
    divt[idx] = __builtin_amdgcn_exp2f(-0.0259525590f * (float)idx);
  __syncthreads();

  // xh tile: x[b][h*64+d][s] + pe; pe row = h*64 + s/16, col = (s%16)*64 + d
  {
    const int row = tid & 127;
    const int sg  = s0 + row;
    const int dbase = (tid >> 7) * 32;           // even
    const float* xp = x + ((size_t)b * E_ + h * D_) * 1024 + sg;
    const float prowf = (float)(h * 64 + (sg >> 4));
    const int pbase = (sg & 15) << 6;
    float* xhp = xh_ws + ((size_t)(h * B_ + b) * S_ + sg) * D_;
#pragma unroll
    for (int i = 0; i < 32; i += 2) {            // parity known at compile time
      int d0 = dbase + i;
      float ang = prowf * divt[(pbase + d0) >> 1];
      float v0 = xp[(size_t)d0 * 1024] + __sinf(ang);
      float v1 = xp[(size_t)(d0 + 1) * 1024] + __cosf(ang);
      xhp[d0] = v0;  xhp[d0 + 1] = v1;
      xt[row][d0] = f2b(v0);  xt[row][d0 + 1] = f2b(v1);
    }
  }
  __syncthreads();

  const int lane = tid & 31;
  const int wv   = tid >> 5;
  const size_t obase = ((size_t)(h * B_ + b) * S_ + s0) * D_;

  // each wave touches only its own 16 rows of zt -> no barriers needed
  gemm_layer<true >(&xt[0][0], &Wt[0][0][0], bias6[0], &zt[0][0], nullptr, wv, lane);
  gemm_layer<false>(&zt[0][0], &Wt[1][0][0], bias6[1], nullptr, Qg + obase, wv, lane);
  gemm_layer<true >(&xt[0][0], &Wt[2][0][0], bias6[2], &zt[0][0], nullptr, wv, lane);
  gemm_layer<false>(&zt[0][0], &Wt[3][0][0], bias6[3], nullptr, Kg + obase, wv, lane);
  gemm_layer<true >(&xt[0][0], &Wt[4][0][0], bias6[4], &zt[0][0], nullptr, wv, lane);
  gemm_layer<false>(&zt[0][0], &Wt[5][0][0], bias6[5], nullptr, Vg + obase, wv, lane);
}

// ---------------------------------------------------------------------------
// Kernel 2: software-pipelined flash attention + residual.
// grid: H*B*(S/128) blocks of 256 (8 waves x 16 query rows).
// ---------------------------------------------------------------------------
__device__ __forceinline__ void k_stage_async(__bf16 (*Kt)[72], const u16* Kg,
                                              size_t kb, int tid) {
#pragma unroll
  for (int i = 0; i < 4; ++i) {
    int idx = tid + i * 256;
    int e0 = idx << 3;
    async_ld_b128(lds_off(&Kt[e0 >> 6][e0 & 63]), Kg + kb + e0);
  }
}
__device__ __forceinline__ void v_load_regs(const u16* Vg, size_t kb, int tid,
                                            uint2* vreg) {
  const uint2* src = (const uint2*)(Vg + kb);
#pragma unroll
  for (int i = 0; i < 8; ++i) vreg[i] = src[tid + i * 256];
}
__device__ __forceinline__ void v_store_lds(__bf16 (*Vt)[136], int tid,
                                            const uint2* vreg) {
#pragma unroll
  for (int i = 0; i < 8; ++i) {
    int idx = tid + i * 256;
    int e0 = idx << 2;
    int t = e0 >> 6, d0 = e0 & 63;
    uint2 vv = vreg[i];
    Vt[d0 + 0][t] = u2b((u16)(vv.x & 0xffffu));
    Vt[d0 + 1][t] = u2b((u16)(vv.x >> 16));
    Vt[d0 + 2][t] = u2b((u16)(vv.y & 0xffffu));
    Vt[d0 + 3][t] = u2b((u16)(vv.y >> 16));
  }
}

__global__ __launch_bounds__(256) void attn_kernel(
    const u16* __restrict__ Qg, const u16* __restrict__ Kg,
    const u16* __restrict__ Vg, const float* __restrict__ xh_ws,
    float* __restrict__ y_ws) {
  __shared__ alignas(16) __bf16 Qt[128][72];
  __shared__ alignas(16) __bf16 Kt[2][128][72];  // double-buffered key tiles
  __shared__ alignas(16) __bf16 Vt[64][136];     // transposed: [d][t]
  __shared__ alignas(16) __bf16 Pt[128][136];    // P relayout buffer

  const int tid = threadIdx.x;
  const int bid = blockIdx.x;
  const int h = bid >> 6, b = (bid >> 3) & 7, q0 = (bid & 7) << 7;
  const size_t hb = (size_t)(h * B_ + b);
  const size_t qbase = (hb * S_ + q0) * D_;
  const size_t kvbase = hb * S_ * D_;

  // prologue: async Q + K0, V0 global loads into registers
#pragma unroll
  for (int i = 0; i < 4; ++i) {
    int idx = tid + i * 256;
    int e0 = idx << 3;
    async_ld_b128(lds_off(&Qt[e0 >> 6][e0 & 63]), Qg + qbase + e0);
  }
  k_stage_async(Kt[0], Kg, kvbase, tid);
  uint2 vreg[8];
  v_load_regs(Vg, kvbase, tid, vreg);
  wait_async0();
  v_store_lds(Vt, tid, vreg);
  __syncthreads();

  const int lane = tid & 31, wv = tid >> 5;
  const int n16 = lane & 15, half = lane >> 4;

  // Q A-fragments are invariant across all key tiles: cache in registers
  const __bf16* qrow = &Qt[0][0] + (wv * 16 + n16) * 72;
  const v16bf aq0 = loadA(qrow, half * 8);
  const v16bf aq1 = loadA(qrow, 32 + half * 8);

  float m8[8], l8[8];
  v8f o[4];
#pragma unroll
  for (int r = 0; r < 8; ++r) { m8[r] = -3.0e38f; l8[r] = 0.f; }
#pragma unroll
  for (int nd = 0; nd < 4; ++nd) o[nd] = zero8();

  // scores scaled into exp2 domain: log2(e)/sqrt(E)
  const float SCL = 0.0450842200f;

  for (int kt = 0; kt < 8; ++kt) {
    const int cur = kt & 1;
    if (kt + 1 < 8) {   // overlap next-tile staging with this tile's compute
      size_t kbn = kvbase + (size_t)(kt + 1) * 128 * D_;
      k_stage_async(Kt[cur ^ 1], Kg, kbn, tid);
      v_load_regs(Vg, kbn, tid, vreg);
    }
    if (kt + 2 < 8) {   // pull V(kt+2) toward L2 (global_prefetch_b8)
      const char* nv = (const char*)(Vg + kvbase + (size_t)(kt + 2) * 128 * D_);
      __builtin_prefetch(nv + tid * 64, 0, 1);
    }

    // scores: this wave's 16 rows x 128 keys on Kt[cur]
    const __bf16* ktile = &Kt[cur][0][0];
    v8f sc[8];
#pragma unroll
    for (int j = 0; j < 8; ++j) {
      const __bf16* krow = ktile + (j * 16 + n16) * 72;
      v8f acc = zero8();
      acc = wmma_bf16(aq0, loadB(krow, half * 8), acc);
      acc = wmma_bf16(aq1, loadB(krow, 32 + half * 8), acc);
      sc[j] = acc * SCL;
    }

    // online softmax (exp2 domain); row r <-> global row r + 8*half
#pragma unroll
    for (int r = 0; r < 8; ++r) {
      float mx = sc[0][r];
#pragma unroll
      for (int j = 1; j < 8; ++j) mx = fmaxf(mx, sc[j][r]);
#pragma unroll
      for (int msk = 1; msk <= 8; msk <<= 1) mx = fmaxf(mx, __shfl_xor(mx, msk, 32));
      float newm = fmaxf(m8[r], mx);
      float corr = __builtin_amdgcn_exp2f(m8[r] - newm);
      float rs = 0.f;
      __bf16* prow = &Pt[0][0] + (wv * 16 + r + half * 8) * 136;
#pragma unroll
      for (int j = 0; j < 8; ++j) {
        float p = __builtin_amdgcn_exp2f(sc[j][r] - newm);
        rs += p;
        prow[j * 16 + n16] = f2b(p);              // C-layout -> LDS for A reload
      }
#pragma unroll
      for (int msk = 1; msk <= 8; msk <<= 1) rs += __shfl_xor(rs, msk, 32);
      l8[r] = l8[r] * corr + rs;
      m8[r] = newm;
#pragma unroll
      for (int nd = 0; nd < 4; ++nd) o[nd][r] *= corr;
    }

    // O += P(16x128) @ V(128x64)
    const __bf16* parow = &Pt[0][0] + (wv * 16 + n16) * 136;
#pragma unroll
    for (int kc = 0; kc < 4; ++kc) {
      v16bf a = loadA(parow, kc * 32 + half * 8);
#pragma unroll
      for (int nd = 0; nd < 4; ++nd) {
        v16bf bb = loadB(&Vt[0][0] + (nd * 16 + n16) * 136, kc * 32 + half * 8);
        o[nd] = wmma_bf16(a, bb, o[nd]);
      }
    }

    __syncthreads();                  // all waves done reading Vt / Kt[cur]
    if (kt + 1 < 8) {
      wait_async0();                  // K(kt+1) landed in Kt[cur^1]
      v_store_lds(Vt, tid, vreg);     // install V(kt+1)
      __syncthreads();
    }
  }

  // finalize: O/l + residual xh; y stored as (B,S,E) fp32
#pragma unroll
  for (int r = 0; r < 8; ++r) {
    float inv = __builtin_amdgcn_rcpf(l8[r]);
    int sg = q0 + wv * 16 + r + half * 8;
    const float* xr = xh_ws + (hb * S_ + sg) * D_;
    float* yr = y_ws + ((size_t)b * S_ + sg) * E_ + h * D_;
#pragma unroll
    for (int nd = 0; nd < 4; ++nd) {
      int col = nd * 16 + n16;
      yr[col] = o[nd][r] * inv + xr[col];
    }
  }
}

// ---------------------------------------------------------------------------
// Kernel 3: LayerNorm over E, output transposed to (B,E,S). grid: B*S blocks.
// ---------------------------------------------------------------------------
__global__ __launch_bounds__(256) void ln_kernel(
    const float* __restrict__ y_ws, const float* __restrict__ gamma,
    const float* __restrict__ beta, float* __restrict__ out) {
  __shared__ float r1[256], r2[256];
  const int tid = threadIdx.x;
  const int bs = blockIdx.x;           // b*1024 + s
  const int b = bs >> 10, s = bs & 1023;
  const float* row = y_ws + (size_t)bs * E_;
  float v[4], sum = 0.f, sq = 0.f;
#pragma unroll
  for (int i = 0; i < 4; ++i) {
    v[i] = row[tid + i * 256];
    sum += v[i];
    sq  += v[i] * v[i];
  }
  r1[tid] = sum; r2[tid] = sq;
  __syncthreads();
  for (int off = 128; off > 0; off >>= 1) {
    if (tid < off) { r1[tid] += r1[tid + off]; r2[tid] += r2[tid + off]; }
    __syncthreads();
  }
  float mu  = r1[0] * (1.0f / 1024.0f);
  float var = r2[0] * (1.0f / 1024.0f) - mu * mu;
  float inv = rsqrtf(var + 1e-5f);
#pragma unroll
  for (int i = 0; i < 4; ++i) {
    int e = tid + i * 256;
    out[((size_t)b * E_ + e) * S_ + s] = (v[i] - mu) * inv * gamma[e] + beta[e];
  }
}

// ---------------------------------------------------------------------------
extern "C" void kernel_launch(void* const* d_in, const int* in_sizes, int n_in,
                              void* d_out, int out_size, void* d_ws,
                              size_t ws_size, hipStream_t stream) {
  (void)in_sizes; (void)n_in; (void)out_size; (void)ws_size;
  const float* x   = (const float*)d_in[0];
  const float* Wq1 = (const float*)d_in[1];
  const float* bq1 = (const float*)d_in[2];
  const float* Wq2 = (const float*)d_in[3];
  const float* bq2 = (const float*)d_in[4];
  const float* Wk1 = (const float*)d_in[5];
  const float* bk1 = (const float*)d_in[6];
  const float* Wk2 = (const float*)d_in[7];
  const float* bk2 = (const float*)d_in[8];
  const float* Wv1 = (const float*)d_in[9];
  const float* bv1 = (const float*)d_in[10];
  const float* Wv2 = (const float*)d_in[11];
  const float* bv2 = (const float*)d_in[12];
  const float* gamma = (const float*)d_in[13];
  const float* beta  = (const float*)d_in[14];

  // workspace layout: xh f32 (32MB) | Q,K,V bf16 (3x16MB) | y f32 (32MB) = 112MB
  const size_t HBSD = (size_t)H_ * B_ * S_ * D_;   // 8,388,608
  float* xh_ws = (float*)d_ws;
  u16* Qg = (u16*)(xh_ws + HBSD);
  u16* Kg = Qg + HBSD;
  u16* Vg = Kg + HBSD;
  float* y_ws = (float*)(Vg + HBSD);

  qkv_posenc_kernel<<<H_ * B_ * (S_ / 128), 256, 0, stream>>>(
      x, Wq1, bq1, Wq2, bq2, Wk1, bk1, Wk2, bk2, Wv1, bv1, Wv2, bv2,
      xh_ws, Qg, Kg, Vg);
  attn_kernel<<<H_ * B_ * (S_ / 128), 256, 0, stream>>>(Qg, Kg, Vg, xh_ws, y_ws);
  ln_kernel<<<B_ * S_, 256, 0, stream>>>(y_ws, gamma, beta, (float*)d_out);
}
